// InfoNCELoss_60387240181909
// MI455X (gfx1250) — matrix operand order repaired and checked by reading, not compiled, not run on hardware
//
#include <hip/hip_runtime.h>

// ---------------------------------------------------------------------------
// InfoNCE loss (cosine sim + streaming logsumexp) for MI455X / gfx1250.
// batch[8192,512] fp32 -> scalar fp32.
//   Pass 1: row-normalize, split fp32 -> bf16 hi + bf16 lo  (ws)
//   Pass 2: fused Gram-matrix (WMMA bf16, split-3) + online logsumexp + pos
//   Pass 3: mean reduce
// ---------------------------------------------------------------------------

typedef __attribute__((ext_vector_type(16))) __bf16        v16bf;
typedef __attribute__((ext_vector_type(8)))  float         v8f;
typedef __attribute__((ext_vector_type(4)))  unsigned int  u32x4;
typedef __attribute__((ext_vector_type(4)))  float         f32x4;

union FragBF { u32x4 u[2]; v16bf v; };   // 8 VGPRs = one 16x32 bf16 fragment

#define D_DIM    512
#define NEG_FILL (-9.0e15f)
#define INV_T    (1.0f / 0.07f)

// fp32 -> bf16 (round to nearest even), and back, via raw bits
__device__ __forceinline__ unsigned short f2bf(float f) {
    unsigned int u = __float_as_uint(f);
    u += 0x7FFFu + ((u >> 16) & 1u);
    return (unsigned short)(u >> 16);
}
__device__ __forceinline__ float bf2f(unsigned short h) {
    return __uint_as_float(((unsigned int)h) << 16);
}

// ---------------------------------------------------------------------------
// Pass 1: one wave32 per row. 512 cols -> 16 per lane (4x float4 loads).
// ---------------------------------------------------------------------------
__global__ __launch_bounds__(256)
void nce_normalize_split(const float* __restrict__ x,
                         unsigned short* __restrict__ xh,
                         unsigned short* __restrict__ xl, int n) {
    const int wave = threadIdx.x >> 5;
    const int lane = threadIdx.x & 31;
    const int row  = blockIdx.x * 8 + wave;
    if (row >= n) return;

    const f32x4* xr = (const f32x4*)(x + (size_t)row * D_DIM);
    f32x4 v[4];
#pragma unroll
    for (int i = 0; i < 4; ++i) v[i] = xr[lane * 4 + i];

    float ss = 0.f;
#pragma unroll
    for (int i = 0; i < 4; ++i)
        ss += v[i].x * v[i].x + v[i].y * v[i].y + v[i].z * v[i].z + v[i].w * v[i].w;
#pragma unroll
    for (int off = 16; off > 0; off >>= 1) ss += __shfl_xor(ss, off, 32);

    const float scale = 1.0f / fmaxf(sqrtf(ss), 1e-8f);

    unsigned int ph[8], pl[8];             // 16 bf16 hi + 16 bf16 lo, packed
#pragma unroll
    for (int i = 0; i < 4; ++i) {
        float e[4] = { v[i].x * scale, v[i].y * scale, v[i].z * scale, v[i].w * scale };
#pragma unroll
        for (int j = 0; j < 2; ++j) {
            unsigned short h0 = f2bf(e[2 * j]),     h1 = f2bf(e[2 * j + 1]);
            unsigned short l0 = f2bf(e[2 * j]     - bf2f(h0));
            unsigned short l1 = f2bf(e[2 * j + 1] - bf2f(h1));
            ph[i * 2 + j] = (unsigned int)h0 | ((unsigned int)h1 << 16);
            pl[i * 2 + j] = (unsigned int)l0 | ((unsigned int)l1 << 16);
        }
    }
    u32x4* oh = (u32x4*)(xh + (size_t)row * D_DIM + lane * 16);
    u32x4* ol = (u32x4*)(xl + (size_t)row * D_DIM + lane * 16);
    oh[0] = u32x4{ph[0], ph[1], ph[2], ph[3]};
    oh[1] = u32x4{ph[4], ph[5], ph[6], ph[7]};
    ol[0] = u32x4{pl[0], pl[1], pl[2], pl[3]};
    ol[1] = u32x4{pl[4], pl[5], pl[6], pl[7]};
}

// ---------------------------------------------------------------------------
// Pass 2: 4 waves / block, 16 rows per wave (BM=64). Columns in BN=64 blocks,
// B chunks (64 cols x 32 K, hi+lo) staged in LDS. Split-bf16: hh + hl + lh.
// Online (max, sumexp) per C register; diag mask + positive in epilogue.
// ---------------------------------------------------------------------------
#define BS_STRIDE 40   // ushorts per staged column row: 80B, 16B-aligned, no bank conflicts

__global__ __launch_bounds__(128)
void nce_gram_lse(const unsigned short* __restrict__ xh,
                  const unsigned short* __restrict__ xl,
                  float* __restrict__ row_out, int n) {
    __shared__ unsigned short BsH[64 * BS_STRIDE];
    __shared__ unsigned short BsL[64 * BS_STRIDE];

    const int tid  = threadIdx.x;
    const int lane = tid & 31;
    const int wave = tid >> 5;
    const int mrow = lane & 15;          // A row / B col within fragment
    const int hh   = lane >> 4;          // lane half (ISA fragment layout)
    const int gm0  = blockIdx.x * 64 + wave * 16;

    float mx[8], sm[8], ps[8];
#pragma unroll
    for (int r = 0; r < 8; ++r) { mx[r] = -__builtin_inff(); sm[r] = 0.f; ps[r] = 0.f; }

    const unsigned short* aH = xh + (size_t)(gm0 + mrow) * D_DIM;
    const unsigned short* aL = xl + (size_t)(gm0 + mrow) * D_DIM;
    const int kbaseA = hh * 16;          // A: halves own K {0..15} / {16..31}

    for (int j0 = 0; j0 < n; j0 += 64) {
        v8f acc[4];
#pragma unroll
        for (int t = 0; t < 4; ++t) acc[t] = v8f{0.f,0.f,0.f,0.f,0.f,0.f,0.f,0.f};

        for (int k0 = 0; k0 < D_DIM; k0 += 32) {
            __syncthreads();
            {   // cooperative stage: 128 threads x (2+2) 16B vectors
                const int c  = tid >> 1;
                const int ko = (tid & 1) << 4;
                const unsigned short* gH = xh + (size_t)(j0 + c) * D_DIM + k0 + ko;
                const unsigned short* gL = xl + (size_t)(j0 + c) * D_DIM + k0 + ko;
                u32x4* sH = (u32x4*)(BsH + c * BS_STRIDE + ko);
                u32x4* sL = (u32x4*)(BsL + c * BS_STRIDE + ko);
                sH[0] = ((const u32x4*)gH)[0];  sH[1] = ((const u32x4*)gH)[1];
                sL[0] = ((const u32x4*)gL)[0];  sL[1] = ((const u32x4*)gL)[1];
                if (k0 + 32 < D_DIM) {           // global_prefetch_b8 next K chunk
                    __builtin_prefetch(gH + 32, 0, 0);
                    __builtin_prefetch(gL + 32, 0, 0);
                }
            }
            __syncthreads();

            // A fragments (contiguous K order, matching B): K kbase..+15
            FragBF aHf, aLf;
            {
                const u32x4* pH = (const u32x4*)(aH + k0 + kbaseA);
                const u32x4* pL = (const u32x4*)(aL + k0 + kbaseA);
                aHf.u[0] = pH[0];  aHf.u[1] = pH[1];
                aLf.u[0] = pL[0];  aLf.u[1] = pL[1];
            }

#pragma unroll
            for (int t = 0; t < 4; ++t) {
                // B fragment (32x16): lanes 0-15 K=0..15, lanes 16-31 K=16..31
                const int cl = t * 16 + mrow;
                const int kb = hh * 16;
                FragBF bHf, bLf;
                const u32x4* bH = (const u32x4*)(BsH + cl * BS_STRIDE + kb);
                const u32x4* bL = (const u32x4*)(BsL + cl * BS_STRIDE + kb);
                bHf.u[0] = bH[0];  bHf.u[1] = bH[1];
                bLf.u[0] = bL[0];  bLf.u[1] = bL[1];

                acc[t] = __builtin_amdgcn_wmma_f32_16x16x32_bf16(
                            false, aHf.v, false, bHf.v, (short)0, acc[t], false, false);
                acc[t] = __builtin_amdgcn_wmma_f32_16x16x32_bf16(
                            false, aHf.v, false, bLf.v, (short)0, acc[t], false, false);
                acc[t] = __builtin_amdgcn_wmma_f32_16x16x32_bf16(
                            false, aLf.v, false, bHf.v, (short)0, acc[t], false, false);
            }
        }

        // epilogue: mask diag, /T, positive capture, online logsumexp
#pragma unroll
        for (int t = 0; t < 4; ++t) {
            const int gn = j0 + t * 16 + mrow;
#pragma unroll
            for (int r = 0; r < 8; ++r) {
                const int gm = gm0 + hh * 8 + r;      // C layout: VGPR r -> row (half*8+r)
                float v = acc[t][r];
                if (gm == gn) v = NEG_FILL;
                const float logit = v * INV_T;
                if (gn == (gm ^ 4096)) ps[r] += logit; // (i - N/2) mod N == i XOR N/2
                const float mn = fmaxf(mx[r], logit);
                sm[r] = sm[r] * __expf(mx[r] - mn) + __expf(logit - mn);
                mx[r] = mn;
            }
        }
    }

    // combine the 16 lanes of each half (rows 0-7 in lanes 0-15, 8-15 in 16-31)
#pragma unroll
    for (int r = 0; r < 8; ++r) {
        float m = mx[r], s = sm[r], p = ps[r];
#pragma unroll
        for (int off = 1; off < 16; off <<= 1) {
            const float mo = __shfl_xor(m, off, 32);
            const float so = __shfl_xor(s, off, 32);
            p += __shfl_xor(p, off, 32);
            const float mn = fmaxf(m, mo);
            s = s * __expf(m - mn) + so * __expf(mo - mn);
            m = mn;
        }
        mx[r] = m; sm[r] = s; ps[r] = p;
    }
    if (mrow == 0) {
#pragma unroll
        for (int r = 0; r < 8; ++r)
            row_out[gm0 + hh * 8 + r] = -ps[r] + mx[r] + logf(sm[r]);
    }
}

// ---------------------------------------------------------------------------
// Pass 3: mean of n per-row losses -> d_out[0]
// ---------------------------------------------------------------------------
__global__ __launch_bounds__(256)
void nce_mean(const float* __restrict__ row_out, float* __restrict__ out, int n) {
    __shared__ float buf[256];
    float s = 0.f;
    for (int i = threadIdx.x; i < n; i += 256) s += row_out[i];
    buf[threadIdx.x] = s;
    __syncthreads();
    for (int st = 128; st > 0; st >>= 1) {
        if (threadIdx.x < st) buf[threadIdx.x] += buf[threadIdx.x + st];
        __syncthreads();
    }
    if (threadIdx.x == 0) out[0] = buf[0] / (float)n;
}

// ---------------------------------------------------------------------------
extern "C" void kernel_launch(void* const* d_in, const int* in_sizes, int n_in,
                              void* d_out, int out_size, void* d_ws, size_t ws_size,
                              hipStream_t stream) {
    const float* batch = (const float*)d_in[0];
    // d_in[1] (targets) is unused by the reference.
    const int n = in_sizes[0] / D_DIM;                 // 8192

    unsigned short* xh = (unsigned short*)d_ws;                 // n*512 bf16 hi
    unsigned short* xl = xh + (size_t)n * D_DIM;                // n*512 bf16 lo
    float* row_out     = (float*)(xl + (size_t)n * D_DIM);      // n floats

    nce_normalize_split<<<n / 8, 256, 0, stream>>>(batch, xh, xl, n);
    nce_gram_lse<<<n / 64, 128, 0, stream>>>(xh, xl, row_out, n);
    nce_mean<<<1, 256, 0, stream>>>(row_out, (float*)d_out, n);
}
